// SparseRouter_53034256171157
// MI455X (gfx1250) — matrix operand order repaired and checked by reading, not compiled, not run on hardware
//
#include <hip/hip_runtime.h>
#include <math.h>

#define T_TOK 8192
#define HID   1024
#define HID2  2048
#define NEXP  8
#define MT    32            // tokens per FFN tile
#define XSTR  (HID + 8)     // padded LDS strides (16B-aligned rows, bank-conflict spread)
#define HSTR  (HID2 + 8)

typedef __attribute__((ext_vector_type(16))) __bf16 v16bf;
typedef __attribute__((ext_vector_type(8)))  float  v8f;

union Frag { v16bf v; uint4 q[2]; };

__device__ __forceinline__ unsigned short f2bf(float f) {
  union { float f; unsigned u; } v; v.f = f;
  unsigned u = v.u;
  unsigned r = u + 0x7FFFu + ((u >> 16) & 1u);   // round-to-nearest-even
  return (unsigned short)(r >> 16);
}

__global__ void moe_init(int* counts) {
  if (threadIdx.x < NEXP) counts[threadIdx.x] = 0;
}

__global__ void cvt_bf16(const float* __restrict__ src, unsigned short* __restrict__ dst, int n4) {
  int i = blockIdx.x * blockDim.x + threadIdx.x;
  int stride = gridDim.x * blockDim.x;
  for (; i < n4; i += stride) {
    float4 f = ((const float4*)src)[i];
    ushort4 o;
    o.x = f2bf(f.x); o.y = f2bf(f.y); o.z = f2bf(f.z); o.w = f2bf(f.w);
    ((ushort4*)dst)[i] = o;
  }
}

// One wave32 per token: logits for all 8 experts, top-2 + softmax, build expert lists.
__global__ void router_kernel(const float* __restrict__ x, const float* __restrict__ rw,
                              int* counts, int* perm, float* gates) {
  int wave = threadIdx.x >> 5;
  int lane = threadIdx.x & 31;
  int t = blockIdx.x * 8 + wave;
  const float* xr = x + (size_t)t * HID;
  float acc[NEXP];
#pragma unroll
  for (int e = 0; e < NEXP; ++e) acc[e] = 0.f;
  for (int h = lane; h < HID; h += 32) {
    float xv = xr[h];
#pragma unroll
    for (int e = 0; e < NEXP; ++e) acc[e] = fmaf(xv, rw[e * HID + h], acc[e]);
  }
#pragma unroll
  for (int e = 0; e < NEXP; ++e)
    for (int m = 16; m >= 1; m >>= 1) acc[e] += __shfl_xor(acc[e], m, 32);
  if (lane == 0) {
    int i0 = 0; float v0 = acc[0];
#pragma unroll
    for (int e = 1; e < NEXP; ++e) if (acc[e] > v0) { v0 = acc[e]; i0 = e; }
    int i1 = -1; float v1 = -3.4e38f;
#pragma unroll
    for (int e = 0; e < NEXP; ++e) if (e != i0 && acc[e] > v1) { v1 = acc[e]; i1 = e; }
    float ee = expf(v1 - v0);            // v1 <= v0, stable 2-way softmax
    float inv = 1.f / (1.f + ee);
    int p0 = atomicAdd(&counts[i0], 1);
    perm [i0 * T_TOK + p0] = (t << 1);
    gates[i0 * T_TOK + p0] = inv;
    int p1 = atomicAdd(&counts[i1], 1);
    perm [i1 * T_TOK + p1] = (t << 1) | 1;
    gates[i1 * T_TOK + p1] = ee * inv;
  }
}

// Load A fragments (two 16-row M-subtiles) from LDS at K offset kk.
// 16-bit A layout: lane<16 -> K {kk..kk+7, kk+16..kk+23}; lane>=16 -> +8.
__device__ __forceinline__ void load_a(const unsigned short* __restrict__ base, int strd,
                                       int lmod, int lhalf, int kk, Frag& a0, Frag& a1) {
  const unsigned short* ar0 = base + lmod * strd + kk + lhalf * 8;
  a0.q[0] = *(const uint4*)(ar0);
  a0.q[1] = *(const uint4*)(ar0 + 16);
  const unsigned short* ar1 = base + (16 + lmod) * strd + kk + lhalf * 8;
  a1.q[0] = *(const uint4*)(ar1);
  a1.q[1] = *(const uint4*)(ar1 + 16);
}

// Per (expert, 32-token tile): y = GELU(X W1^T) W2^T, scaled by gate.
// slot0 rows -> out (every token covered exactly once), slot1 rows -> partial.
__global__ void __launch_bounds__(512, 1)
moe_ffn(const float* __restrict__ x,
        const unsigned short* __restrict__ w1b,
        const unsigned short* __restrict__ w2b,
        const int* __restrict__ counts,
        const int* __restrict__ perm,
        const float* __restrict__ gates,
        float* __restrict__ out, float* __restrict__ partial) {
  __shared__ unsigned short Xs[MT * XSTR];   // 66 KB
  __shared__ unsigned short Hs[MT * HSTR];   // 131.6 KB
  __shared__ int   toks[MT];
  __shared__ int   slot[MT];
  __shared__ float gts[MT];

  int e = blockIdx.y;
  int cnt = counts[e];
  int base = blockIdx.x * MT;
  if (base >= cnt) return;                    // uniform early-exit
  int valid = cnt - base; if (valid > MT) valid = MT;

  int tid = threadIdx.x;
  if (tid < MT) {
    if (tid < valid) {
      int p = perm[e * T_TOK + base + tid];
      toks[tid] = p >> 1;
      slot[tid] = p & 1;
      gts [tid] = gates[e * T_TOK + base + tid];
    } else { toks[tid] = 0; slot[tid] = 0; gts[tid] = 0.f; }
  }
  __syncthreads();

  // Gather token rows, fp32 -> bf16 into LDS (coalesced float4 loads)
  for (int i = tid; i < MT * (HID / 4); i += 512) {
    int r = i >> 8;                 // HID/4 == 256
    int c = (i & 255) << 2;
    float4 f = *(const float4*)(x + (size_t)toks[r] * HID + c);
    ushort4 o;
    o.x = f2bf(f.x); o.y = f2bf(f.y); o.z = f2bf(f.z); o.w = f2bf(f.w);
    *(ushort4*)(&Xs[r * XSTR + c]) = o;
  }
  __syncthreads();

  int wv = tid >> 5, lane = tid & 31;
  int lmod = lane & 15, lhalf = lane >> 4;

  // ---------------- GEMM1: h[32,2048] = GELU(X @ W1^T), W1[e] is [2048,1024]
  // Register-blocked: 4 N-tiles x 2 M-subtiles of accumulators held across K.
  const unsigned short* w1e = w1b + (size_t)e * HID2 * HID;
  for (int nc = 0; nc < 2; ++nc) {
    int nb0 = (wv * 8 + nc * 4) * 16;                    // first of 4 N-tiles
    const unsigned short* wbase = w1e + (size_t)(nb0 + lmod) * HID + lhalf * 8;
    v8f acc[4][2];
#pragma unroll
    for (int j = 0; j < 4; ++j) { acc[j][0] = (v8f){}; acc[j][1] = (v8f){}; }
#pragma unroll 1
    for (int kk = 0; kk < HID; kk += 32) {
      Frag a0, a1;
      load_a(Xs, XSTR, lmod, lhalf, kk, a0, a1);
      Frag b[4];
#pragma unroll
      for (int j = 0; j < 4; ++j) {                      // 8 global_load_b128, clause-able
        const unsigned short* wr = wbase + (size_t)j * 16 * HID + kk;
        b[j].q[0] = *(const uint4*)(wr);
        b[j].q[1] = *(const uint4*)(wr + 16);
      }
#pragma unroll
      for (int j = 0; j < 4; ++j) {                      // 8 WMMAs per K-step
        acc[j][0] = __builtin_amdgcn_wmma_f32_16x16x32_bf16(false, a0.v, false, b[j].v, (short)0, acc[j][0], false, false);
        acc[j][1] = __builtin_amdgcn_wmma_f32_16x16x32_bf16(false, a1.v, false, b[j].v, (short)0, acc[j][1], false, false);
      }
    }
#pragma unroll
    for (int j = 0; j < 4; ++j) {                        // exact erf-GELU, write bf16 h
      int nb = nb0 + j * 16;
#pragma unroll
      for (int i = 0; i < 8; ++i) {
        int m0 = i + lhalf * 8;
        float g0 = 0.5f * acc[j][0][i] * (1.f + erff(acc[j][0][i] * 0.70710678f));
        Hs[m0 * HSTR + nb + lmod] = f2bf(g0);
        float g1 = 0.5f * acc[j][1][i] * (1.f + erff(acc[j][1][i] * 0.70710678f));
        Hs[(16 + m0) * HSTR + nb + lmod] = f2bf(g1);
      }
    }
  }
  __syncthreads();

  // ---------------- GEMM2: y[32,1024] = h @ W2^T, W2[e] is [1024,2048]
  const unsigned short* w2e = w2b + (size_t)e * HID * HID2;
  {
    int nb0 = (wv * 4) * 16;                             // wave's 4 output N-tiles
    const unsigned short* wbase = w2e + (size_t)(nb0 + lmod) * HID2 + lhalf * 8;
    v8f acc[4][2];
#pragma unroll
    for (int j = 0; j < 4; ++j) { acc[j][0] = (v8f){}; acc[j][1] = (v8f){}; }
#pragma unroll 1
    for (int kk = 0; kk < HID2; kk += 32) {
      Frag a0, a1;
      load_a(Hs, HSTR, lmod, lhalf, kk, a0, a1);
      Frag b[4];
#pragma unroll
      for (int j = 0; j < 4; ++j) {
        const unsigned short* wr = wbase + (size_t)j * 16 * HID2 + kk;
        b[j].q[0] = *(const uint4*)(wr);
        b[j].q[1] = *(const uint4*)(wr + 16);
      }
#pragma unroll
      for (int j = 0; j < 4; ++j) {
        acc[j][0] = __builtin_amdgcn_wmma_f32_16x16x32_bf16(false, a0.v, false, b[j].v, (short)0, acc[j][0], false, false);
        acc[j][1] = __builtin_amdgcn_wmma_f32_16x16x32_bf16(false, a1.v, false, b[j].v, (short)0, acc[j][1], false, false);
      }
    }
    // C layout: lanes 0-15 N=lane, M=i; lanes 16-31 N=lane-16, M=8+i
#pragma unroll
    for (int j = 0; j < 4; ++j) {
      int col = nb0 + j * 16 + lmod;
#pragma unroll
      for (int i = 0; i < 8; ++i) {
        int m0 = i + lhalf * 8;
        int m1 = 16 + m0;
        if (m0 < valid) {
          float* dst = (slot[m0] ? partial : out) + (size_t)toks[m0] * HID + col;
          *dst = acc[j][0][i] * gts[m0];
        }
        if (m1 < valid) {
          float* dst = (slot[m1] ? partial : out) + (size_t)toks[m1] * HID + col;
          *dst = acc[j][1][i] * gts[m1];
        }
      }
    }
  }
}

__global__ void combine_kernel(float* __restrict__ out, const float* __restrict__ partial, int n4) {
  int i = blockIdx.x * blockDim.x + threadIdx.x;
  if (i < n4) {
    float4 a = ((const float4*)out)[i];
    float4 b = ((const float4*)partial)[i];
    a.x += b.x; a.y += b.y; a.z += b.z; a.w += b.w;
    ((float4*)out)[i] = a;
  }
}

extern "C" void kernel_launch(void* const* d_in, const int* in_sizes, int n_in,
                              void* d_out, int out_size, void* d_ws, size_t ws_size,
                              hipStream_t stream) {
  (void)in_sizes; (void)n_in; (void)out_size; (void)ws_size;
  const float* x  = (const float*)d_in[0];   // [T, H]
  const float* rw = (const float*)d_in[1];   // [E, H]
  const float* w1 = (const float*)d_in[2];   // [E, 2H, H]
  const float* w2 = (const float*)d_in[3];   // [E, H, 2H]
  float* out = (float*)d_out;                // [T, H]

  // Workspace layout (~97 MB): bf16 w1 | bf16 w2 | partial fp32 | counts | perm | gates
  size_t wElems = (size_t)NEXP * HID2 * HID;           // 16,777,216 per tensor
  unsigned short* w1b = (unsigned short*)d_ws;
  unsigned short* w2b = w1b + wElems;
  float* partial = (float*)(w2b + wElems);
  int*   counts  = (int*)(partial + (size_t)T_TOK * HID);
  int*   perm    = counts + 16;
  float* gates   = (float*)(perm + NEXP * T_TOK);

  moe_init<<<1, 32, 0, stream>>>(counts);
  int n4 = (int)(wElems / 4);
  cvt_bf16<<<2048, 256, 0, stream>>>(w1, w1b, n4);
  cvt_bf16<<<2048, 256, 0, stream>>>(w2, w2b, n4);
  router_kernel<<<T_TOK / 8, 256, 0, stream>>>(x, rw, counts, perm, gates);
  moe_ffn<<<dim3(T_TOK / MT, NEXP), 512, 0, stream>>>(x, w1b, w2b, counts, perm, gates, out, partial);
  combine_kernel<<<(T_TOK * HID / 4 + 255) / 256, 256, 0, stream>>>(out, partial, T_TOK * HID / 4);
}
